// FixedLegacyGATModel_57028575756675
// MI455X (gfx1250) — matrix operand order repaired and checked by reading, not compiled, not run on hardware
//
#include <hip/hip_runtime.h>
#include <hip/hip_bf16.h>

typedef __attribute__((ext_vector_type(16))) _Float16 v16h;
typedef __attribute__((ext_vector_type(8)))  float    v8f;

#define BATCH 256
#define NNODE 100
#define HDIM  128
#define NHEAD 8
#define HD    16

// ---- WMMA operand gathers (layouts per CDNA5 ISA 7.12.2, 16-bit 16x32 A / 32x16 B) ----
__device__ __forceinline__ v16h load_A_f32(const float* __restrict__ base, int ld, int lane) {
  v16h a;
  int m = lane & 15, half = (lane >> 4) & 1;
#pragma unroll
  for (int j = 0; j < 16; ++j) {
    int k = (j & 1) + 2 * ((j >> 1) & 3) + 8 * half + 16 * (j >> 3);
    a[j] = (_Float16)base[m * ld + k];
  }
  return a;
}
__device__ __forceinline__ v16h load_B_f32(const float* __restrict__ base, int ld, int lane) {
  v16h b;
  int n = lane & 15, half = (lane >> 4) & 1;
#pragma unroll
  for (int j = 0; j < 16; ++j) {
    int k = (j & 7) + 8 * half + 16 * (j >> 3);
    b[j] = (_Float16)base[k * ld + n];
  }
  return b;
}

// Packed-B layout: weight w (128x128) -> 32 tiles (kt in 0..3, nt in 0..7),
// tile t = kt*8+nt, lane's 16 halves contiguous at out[(t*32+lane)*16].
#define WPACK_STRIDE 16384  // halves per packed 128x128 weight

// One wave computes a 16x16 tile of (16x128 A) @ (128x128 W) with packed-B weights.
// wave w owns output columns [16w, 16w+16).
__device__ __forceinline__ void gemm16_wmma_p(const float* __restrict__ A,
                                              const _Float16* __restrict__ Wp,
                                              float* __restrict__ out,
                                              int lane, int wave) {
  v8f acc = {};
#pragma unroll
  for (int kt = 0; kt < 4; ++kt) {
    v16h a = load_A_f32(A + kt * 32, 128, lane);
    v16h b = *(const v16h*)(Wp + ((size_t)(kt * 8 + wave) * 32 + lane) * 16);
    acc = __builtin_amdgcn_wmma_f32_16x16x32_f16(false, a, false, b, (short)0, acc, false, false);
  }
  int n = wave * 16 + (lane & 15);
  int mb = 8 * ((lane >> 4) & 1);
#pragma unroll
  for (int e = 0; e < 8; ++e) out[(mb + e) * 128 + n] = acc[e];
}

// ---- Kernel 0: convert 7 weight matrices f32 -> f16 packed WMMA-B layout ----
__global__ __launch_bounds__(32) void vrp_pack_weights(
    const float* __restrict__ Wk, const float* __restrict__ Wv,
    const float* __restrict__ Wkp, const float* __restrict__ Wfc1,
    const float* __restrict__ Wfc, const float* __restrict__ Wq,
    const float* __restrict__ Wo, _Float16* __restrict__ out) {
  int w = blockIdx.x;   // 0..6
  int t = blockIdx.y;   // 0..31
  int lane = threadIdx.x;
  const float* W;
  switch (w) {
    case 0: W = Wk;   break;
    case 1: W = Wv;   break;
    case 2: W = Wkp;  break;
    case 3: W = Wfc1; break;
    case 4: W = Wfc;  break;   // rows 0..127 only; cap row handled separately
    case 5: W = Wq;   break;
    default: W = Wo;  break;
  }
  int kt = t >> 3, nt = t & 7;
  v16h b = load_B_f32(W + (size_t)kt * 32 * 128 + nt * 16, 128, lane);
  *(v16h*)(out + (size_t)w * WPACK_STRIDE + ((size_t)t * 32 + lane) * 16) = b;
}

// ---- Kernel 1: K_mha / V_mha / K_ptr / pool_proc precompute (dense WMMA GEMM) ----
__global__ __launch_bounds__(256) void vrp_precompute(
    const float* __restrict__ enc, const float* __restrict__ pool,
    const _Float16* __restrict__ Wpack,
    float* __restrict__ K_out, float* __restrict__ V_out,
    float* __restrict__ Kp_out, float* __restrict__ pool_out) {
  int which = blockIdx.y;       // 0:K 1:V 2:Kptr 3:pool_proc
  int mtile = blockIdx.x;       // 0..1599 (25600 rows / 16)
  int wave = threadIdx.x >> 5;  // 0..7 -> N tile
  int lane = threadIdx.x & 31;

  const float* A; float* O;
  if (which == 0)      { A = enc;  O = K_out;  }
  else if (which == 1) { A = enc;  O = V_out;  }
  else if (which == 2) { A = enc;  O = Kp_out; }
  else { if (mtile >= 16) return; A = pool; O = pool_out; }
  const _Float16* Wp = Wpack + (size_t)which * WPACK_STRIDE;

  const float* Ab = A + (size_t)mtile * 16 * 128;
  v8f acc = {};
#pragma unroll
  for (int kt = 0; kt < 4; ++kt) {
    v16h a = load_A_f32(Ab + kt * 32, 128, lane);
    v16h b = *(const v16h*)(Wp + ((size_t)(kt * 8 + wave) * 32 + lane) * 16);
    acc = __builtin_amdgcn_wmma_f32_16x16x32_f16(false, a, false, b, (short)0, acc, false, false);
  }
  int n = wave * 16 + (lane & 15);
  int mb = 8 * ((lane >> 4) & 1);
#pragma unroll
  for (int e = 0; e < 8; ++e)
    O[((size_t)mtile * 16 + mb + e) * 128 + n] = acc[e];
}

// ---- In-LDS LayerNorm over 128 dims, 16 rows x 16 subthreads ----
__device__ __forceinline__ void layernorm16(float (*buf)[128],
                                            const float* __restrict__ g,
                                            const float* __restrict__ b,
                                            float (*red)[16], int r, int sub) {
  float s = 0.f, s2 = 0.f;
#pragma unroll
  for (int j = 0; j < 8; ++j) { float v = buf[r][sub * 8 + j]; s += v; s2 += v * v; }
  red[r][sub] = s;
  __syncthreads();
  float mean = 0.f;
  for (int t = 0; t < 16; ++t) mean += red[r][t];
  mean *= (1.f / 128.f);
  __syncthreads();
  red[r][sub] = s2;
  __syncthreads();
  float m2 = 0.f;
  for (int t = 0; t < 16; ++t) m2 += red[r][t];
  float var = m2 * (1.f / 128.f) - mean * mean;
  float rstd = rsqrtf(var + 1e-5f);
#pragma unroll
  for (int j = 0; j < 8; ++j) {
    int c = sub * 8 + j;
    buf[r][c] = (buf[r][c] - mean) * rstd * g[c] + b[c];
  }
  __syncthreads();
}

// ---- Kernel 2: persistent 150-step decoder. 16 blocks x 16 batch rows. ----
__global__ __launch_bounds__(256) void vrp_decode(
    const float* __restrict__ enc, const float* __restrict__ capacity,
    const float* __restrict__ demand, const float* __restrict__ W_fc,
    const float* __restrict__ dec_g, const float* __restrict__ dec_b,
    const float* __restrict__ ptr_g, const float* __restrict__ ptr_b,
    const _Float16* __restrict__ Wpack,
    const float* __restrict__ K_mha, const float* __restrict__ V_mha,
    const float* __restrict__ K_ptr, const float* __restrict__ pool_proc,
    const int* __restrict__ n_steps_p, const float* __restrict__ T_p,
    float* __restrict__ actions, float* __restrict__ logps) {
  __shared__ float s_att[16][NHEAD][NNODE];   // attention logits -> probs
  __shared__ float s_decin[16][128];
  __shared__ float s_q[16][128];              // q, then reused as attn-output (pre-Wo)
  __shared__ float s_x[16][128];
  __shared__ float s_inp[16][128];
  __shared__ float s_pool[16][128];
  __shared__ float s_demand[16][NNODE];
  __shared__ float s_mask1[16][NNODE];
  __shared__ int   s_mask[16][NNODE];
  __shared__ float s_comp[16][NNODE];
  __shared__ float s_red[16][16];
  __shared__ float s_cap[16], s_maxcap[16], s_logp[16];
  __shared__ int   s_index[16];

  const int tid = threadIdx.x;
  const int lane = tid & 31, wave = tid >> 5;
  const int r = tid >> 4, sub = tid & 15;      // (row, sub-worker)
  const int bb = blockIdx.x * 16;

  const _Float16* Wfc_p = Wpack + 4 * (size_t)WPACK_STRIDE;
  const _Float16* Wq_p  = Wpack + 5 * (size_t)WPACK_STRIDE;
  const _Float16* Wo_p  = Wpack + 6 * (size_t)WPACK_STRIDE;

  const int n_steps = n_steps_p[0];
  const float T = T_p[0];
  const int stride = n_steps + 2;

  for (int idx = tid; idx < 16 * NNODE; idx += 256) {
    int rr = idx / NNODE, c = idx % NNODE;
    s_demand[rr][c] = demand[(size_t)(bb + rr) * NNODE + c];
    s_mask1[rr][c] = 0.f;
  }
  for (int idx = tid; idx < 16 * 128; idx += 256) {
    int rr = idx >> 7, c = idx & 127;
    s_inp[rr][c]  = enc[(size_t)(bb + rr) * NNODE * 128 + c];       // node 0 embedding
    s_pool[rr][c] = pool_proc[(size_t)(bb + rr) * 128 + c];
  }
  if (tid < 16) {
    s_cap[tid] = capacity[bb + tid];
    s_maxcap[tid] = s_cap[tid];
    s_index[tid] = 0;
    s_logp[tid] = 0.f;
  }
  __syncthreads();

  for (int i = 0; i < n_steps; ++i) {
    // ---- mask1 visit update ----
    if (sub == 0) {
      int idx = s_index[r];
      if (i > 0 && idx != 0) s_mask1[r][idx] = 1.f;
    }
    __syncthreads();

    // ---- mask build + any_feasible / fallback ----
    {
      float cap = s_cap[r];
      int feas_local = 0;
      for (int n = sub; n < NNODE; n += 16) {
        if (n >= 1) {
          int inf = (s_mask1[r][n] > 0.5f) || (s_demand[r][n] > cap);
          s_mask[r][n] = inf;
          feas_local |= !inf;
        }
      }
      s_red[r][sub] = (float)feas_local;
    }
    __syncthreads();
    if (sub == 0) {
      int any = 0;
      for (int t = 0; t < 16; ++t) any |= (s_red[r][t] != 0.f);
      int fallback = !any;
      int m0;
      if (fallback)      m0 = 0;                       // only_depot: depot open
      else if (i == 0)   m0 = 1;
      else               m0 = (s_index[r] == 0) && any;
      s_mask[r][0] = m0;
    }
    __syncthreads();

    // ---- dec_in = [inp,cap] @ W_fc + pool_proc ; LN ----
    gemm16_wmma_p(&s_inp[0][0], Wfc_p, &s_decin[0][0], lane, wave);
    __syncthreads();
    for (int idx = tid; idx < 16 * 128; idx += 256) {
      int rr = idx >> 7, c = idx & 127;
      s_decin[rr][c] += s_cap[rr] * W_fc[128 * 128 + c] + s_pool[rr][c];
    }
    __syncthreads();
    layernorm16(s_decin, dec_g, dec_b, s_red, r, sub);

    // ---- q = dec_in @ Wq ----
    gemm16_wmma_p(&s_decin[0][0], Wq_p, &s_q[0][0], lane, wave);
    __syncthreads();

    // ---- masked MHA attention: softmax over n per (row, head) ----
    if (tid < 128) {
      int rr = tid >> 3, h = tid & 7;
      const float* qrow = &s_q[rr][h * HD];
      const float* Kb = K_mha + (size_t)(bb + rr) * NNODE * 128 + h * HD;
      float mx = -__builtin_inff();
      for (int n = 0; n < NNODE; ++n) {
        float a;
        if (s_mask[rr][n]) a = -__builtin_inff();
        else {
          float dot = 0.f;
#pragma unroll
          for (int d = 0; d < HD; ++d) dot += qrow[d] * Kb[(size_t)n * 128 + d];
          a = dot * 0.25f;                              // 1/sqrt(HD)
        }
        s_att[rr][h][n] = a;
        mx = fmaxf(mx, a);
      }
      float sum = 0.f;
      for (int n = 0; n < NNODE; ++n) {
        float e = __expf(s_att[rr][h][n] - mx);
        s_att[rr][h][n] = e;
        sum += e;
      }
      float inv = 1.f / sum;
      for (int n = 0; n < NNODE; ++n) s_att[rr][h][n] *= inv;
    }
    __syncthreads();

    // ---- attn_out = probs @ V (reuse s_q) ----
    {
#pragma unroll
      for (int j = 0; j < 8; ++j) {
        int c = sub * 8 + j;
        int h = c >> 4;
        const float* V = V_mha + (size_t)(bb + r) * NNODE * 128 + c;
        float acc = 0.f;
        for (int n = 0; n < NNODE; ++n) acc += s_att[r][h][n] * V[(size_t)n * 128];
        s_q[r][c] = acc;
      }
    }
    __syncthreads();

    // ---- x = attn_out @ Wo ; LN(ptr) ----
    gemm16_wmma_p(&s_q[0][0], Wo_p, &s_x[0][0], lane, wave);
    __syncthreads();
    layernorm16(s_x, ptr_g, ptr_b, s_red, r, sub);

    // ---- pointer logits comp = x . K_ptr ----
    for (int n = sub; n < NNODE; n += 16) {
      float v;
      if (s_mask[r][n]) v = -__builtin_inff();
      else {
        const float* Kp = K_ptr + ((size_t)(bb + r) * NNODE + n) * 128;
        float d0 = 0.f;
        for (int c = 0; c < 128; ++c) d0 += s_x[r][c] * Kp[c];
        v = d0 * 0.08838834764831845f;                  // 1/sqrt(128)
      }
      s_comp[r][n] = v;
    }
    __syncthreads();

    // ---- selection / bookkeeping: one owner thread per batch row ----
    if (sub == 0) {
      float mx = -__builtin_inff();
      for (int n = 0; n < NNODE; ++n) mx = fmaxf(mx, s_comp[r][n]);
      if (mx == -__builtin_inff()) {                    // all masked -> comp = 0
        for (int n = 0; n < NNODE; ++n) s_comp[r][n] = 0.f;
        mx = 0.f;
      }
      float invT = 1.f / T;
      float sum = 0.f, best = -__builtin_inff();
      int bestn = 0;
      for (int n = 0; n < NNODE; ++n) {
        float e = __expf((s_comp[r][n] - mx) * invT);
        sum += e;
        if (e > best) { best = e; bestn = n; }
      }
      float p = best / sum + 1e-10f;
      float psum = 1.f + (float)NNODE * 1e-10f;
      float lp = __logf(p / psum);
      float visited = 0.f;
      for (int n = 1; n < NNODE; ++n) visited += s_mask1[r][n];
      float done = (visited >= (float)(NNODE - 1)) ? 1.f : 0.f;
      s_logp[r] += lp * (1.f - done);
      s_cap[r] = (bestn == 0) ? s_maxcap[r] : (s_cap[r] - s_demand[r][bestn]);
      s_index[r] = bestn;
      actions[(size_t)(bb + r) * stride + (i + 1)] = (float)bestn;
    }
    __syncthreads();

    // ---- gather new input embedding ----
    {
      int idx = s_index[r];
      const float* src = enc + ((size_t)(bb + r) * NNODE + idx) * 128;
#pragma unroll
      for (int j = 0; j < 8; ++j) {
        int c = sub * 8 + j;
        s_inp[r][c] = src[c];
      }
    }
    __syncthreads();
  }

  if (sub == 0) {
    logps[bb + r] = s_logp[r];
    actions[(size_t)(bb + r) * stride + 0] = 0.f;
  }
}

// ---- Kernel 3: global need_final reduction + last action column ----
__global__ __launch_bounds__(256) void vrp_finalize(float* __restrict__ actions,
                                                    const int* __restrict__ n_steps_p) {
  __shared__ int s_any;
  int ns = n_steps_p[0];
  int stride = ns + 2;
  if (threadIdx.x == 0) s_any = 0;
  __syncthreads();
  float last = actions[(size_t)threadIdx.x * stride + ns];
  if (last != 0.f) atomicOr(&s_any, 1);
  __syncthreads();
  float fin = s_any ? ((last == 0.f) ? -1.f : 0.f) : 0.f;
  actions[(size_t)threadIdx.x * stride + ns + 1] = fin;
}

extern "C" void kernel_launch(void* const* d_in, const int* in_sizes, int n_in,
                              void* d_out, int out_size, void* d_ws, size_t ws_size,
                              hipStream_t stream) {
  (void)in_sizes; (void)n_in; (void)out_size; (void)ws_size;
  const float* enc  = (const float*)d_in[0];
  const float* pool = (const float*)d_in[1];
  const float* cap  = (const float*)d_in[2];
  const float* dem  = (const float*)d_in[3];
  const float* Wfc  = (const float*)d_in[4];
  const float* Wfc1 = (const float*)d_in[5];
  const float* dg   = (const float*)d_in[6];
  const float* db   = (const float*)d_in[7];
  const float* Wq   = (const float*)d_in[8];
  const float* Wk   = (const float*)d_in[9];
  const float* Wv   = (const float*)d_in[10];
  const float* Wo   = (const float*)d_in[11];
  const float* Wkp  = (const float*)d_in[12];
  const float* pg   = (const float*)d_in[13];
  const float* pb   = (const float*)d_in[14];
  const int*   nsp  = (const int*)d_in[15];
  const float* Tp   = (const float*)d_in[16];

  float* ws = (float*)d_ws;
  float* Km = ws;                          // 25600*128 f32
  float* Vm = ws + 3276800;                // 25600*128 f32
  float* Kp = ws + 6553600;                // 25600*128 f32
  float* pp = ws + 9830400;                // 256*128 f32
  _Float16* Wpack = (_Float16*)(ws + 9863168);  // 7 * 16384 f16 (packed WMMA-B)

  float* actions = (float*)d_out;          // 256 x 152
  float* logps   = actions + 256 * 152;    // 256

  dim3 gp(7, 32);
  vrp_pack_weights<<<gp, 32, 0, stream>>>(Wk, Wv, Wkp, Wfc1, Wfc, Wq, Wo, Wpack);
  dim3 g1(1600, 4);
  vrp_precompute<<<g1, 256, 0, stream>>>(enc, pool, Wpack, Km, Vm, Kp, pp);
  vrp_decode<<<16, 256, 0, stream>>>(enc, cap, dem, Wfc, dg, db, pg, pb, Wpack,
                                     Km, Vm, Kp, pp, nsp, Tp, actions, logps);
  vrp_finalize<<<1, 256, 0, stream>>>(actions, nsp);
}
